// MultiHeadedAttention_41635412967971
// MI455X (gfx1250) — matrix operand order repaired and checked by reading, not compiled, not run on hardware
//
#include <hip/hip_runtime.h>

// ---------------- problem constants ----------------
#define HH      12
#define DMODEL  768
#define DKK     64
#define BB      2
#define SS      2048
#define MM      (BB * SS)      // 4096 rows for the big GEMMs

typedef __attribute__((ext_vector_type(16))) _Float16 v16h;
typedef __attribute__((ext_vector_type(8)))  float    v8f;

union FragU { v16h v; uint4 u[2]; };

// ---- WMMA fragment loaders (CDNA5 ISA 7.12.2 layouts) ----
// A (16xK row-major): lane r=lane&15 holds row r; lane hi bit selects the
// 8-element K-subchunks: elems 0..7 = K[k0+hi*8 ..], elems 8..15 = K[k0+16+hi*8 ..]
static __device__ inline v16h load_a_frag(const _Float16* __restrict__ base, int ld,
                                          int row0, int k0, int lane) {
    const int r  = lane & 15;
    const int hi = lane >> 4;
    const _Float16* p = base + (size_t)(row0 + r) * ld + k0 + hi * 8;
    FragU t;
    t.u[0] = *(const uint4*)(p);        // K = k0+hi*8   .. +7
    t.u[1] = *(const uint4*)(p + 16);   // K = k0+16+hi*8.. +7
    return t.v;
}

// B (KxN, sourced from a row-major [N,K] matrix, i.e. B = W^T):
// lane r holds column col0+r: elems 0..7 = K[k0+hi*8 ..], elems 8..15 = K[k0+16+hi*8 ..]
static __device__ inline v16h load_b_frag(const _Float16* __restrict__ base, int ld,
                                          int col0, int k0, int lane) {
    const int r  = lane & 15;
    const int hi = lane >> 4;
    const _Float16* p = base + (size_t)(col0 + r) * ld + k0 + hi * 8;
    FragU t;
    t.u[0] = *(const uint4*)(p);
    t.u[1] = *(const uint4*)(p + 16);
    return t.v;
}

static __device__ inline v8f wmma16(v16h a, v16h b, v8f c) {
    return __builtin_amdgcn_wmma_f32_16x16x32_f16(
        /*neg_a=*/false, a, /*neg_b=*/false, b,
        /*c_mod=*/(short)0, c, /*reuse_a=*/false, /*reuse_b=*/false);
}

// ---------------- fp32 -> f16 conversion ----------------
__global__ __launch_bounds__(256)
void mha_cvt_f32_f16(const float* __restrict__ src, _Float16* __restrict__ dst, int n) {
    for (int i = blockIdx.x * blockDim.x + threadIdx.x; i < n;
         i += gridDim.x * blockDim.x)
        dst[i] = (_Float16)src[i];
}

// ---------------- GEMM: C[M,N] = A[M,K] * W[N,K]^T + bias ----------------
// MODE 0: store f16, permuted to [b,h,s,dk]   (Q/K projections)
// MODE 1: store f16, transposed  [b,h,dk,s]   (V projection, for P*V B-operand)
// MODE 2: store f32 row-major [M,N]           (output projection -> d_out)
template <int MODE>
__global__ __launch_bounds__(256)
void mha_gemm_wmma(const _Float16* __restrict__ A, const _Float16* __restrict__ W,
                   const float* __restrict__ bias, void* __restrict__ outp,
                   int M, int N, int K) {
    const int tid  = threadIdx.x;
    const int lane = tid & 31;
    const int wid  = tid >> 5;
    const int r    = lane & 15;
    const int hi   = lane >> 4;
    const int wm   = wid & 3;                 // 4 row-waves  (32 rows each)
    const int wn   = wid >> 2;                // 2 col-waves  (64 cols each)
    const int row0 = blockIdx.x * 128 + wm * 32;
    const int col0 = blockIdx.y * 128 + wn * 64;

    v8f acc[2][4];
#pragma unroll
    for (int a = 0; a < 2; ++a)
#pragma unroll
        for (int b = 0; b < 4; ++b) acc[a][b] = (v8f){};

    for (int k0 = 0; k0 < K; k0 += 32) {
        v16h af[2];
#pragma unroll
        for (int a = 0; a < 2; ++a)
            af[a] = load_a_frag(A, K, row0 + a * 16, k0, lane);
#pragma unroll
        for (int b = 0; b < 4; ++b) {
            v16h wf = load_b_frag(W, K, col0 + b * 16, k0, lane);
#pragma unroll
            for (int a = 0; a < 2; ++a)
                acc[a][b] = wmma16(af[a], wf, acc[a][b]);
        }
    }

#pragma unroll
    for (int b = 0; b < 4; ++b) {
        const int col = col0 + b * 16 + r;
        const float bv = bias[col];
#pragma unroll
        for (int a = 0; a < 2; ++a) {
#pragma unroll
            for (int i = 0; i < 8; ++i) {
                const int   row = row0 + a * 16 + hi * 8 + i;
                const float val = acc[a][b][i] + bv;
                if (MODE == 2) {
                    ((float*)outp)[(size_t)row * N + col] = val;
                } else {
                    const int bb = row / SS, s = row % SS;
                    const int h = col / DKK, kd = col % DKK;
                    if (MODE == 0)
                        ((_Float16*)outp)[((size_t)(bb * HH + h) * SS + s) * DKK + kd] =
                            (_Float16)val;
                    else
                        ((_Float16*)outp)[((size_t)(bb * HH + h) * DKK + kd) * SS + s] =
                            (_Float16)val;
                }
            }
        }
    }
}

// ---------------- flash attention ----------------
// Q,K layout [bh][S][64] f16 ; VT layout [bh][64][S] f16 ; ctx out [b][s][768] f16.
// grid: (S/128, B*H); 8 waves/block, wave w owns 16 query rows.
__global__ __launch_bounds__(256)
void mha_flash_attn(const _Float16* __restrict__ Qh, const _Float16* __restrict__ Kh,
                    const _Float16* __restrict__ VT, _Float16* __restrict__ ctx) {
    __shared__ _Float16 pshare[8][16][72];   // +8 f16 row pad to spread LDS banks

    const int tid   = threadIdx.x;
    const int lane  = tid & 31;
    const int w     = tid >> 5;
    const int r     = lane & 15;
    const int hi    = lane >> 4;
    const int bh    = blockIdx.y;
    const int qbase = blockIdx.x * 128 + w * 16;

    const _Float16* Qb = Qh + (size_t)bh * SS * DKK;
    const _Float16* Kb = Kh + (size_t)bh * SS * DKK;
    const _Float16* Vb = VT + (size_t)bh * DKK * SS;

    // Q A-fragments: 16 rows x 64 (two K-chunks of 32), resident whole kernel
    v16h qa[2];
#pragma unroll
    for (int kc = 0; kc < 2; ++kc)
        qa[kc] = load_a_frag(Qb, DKK, qbase, kc * 32, lane);

    v8f o[4];
#pragma unroll
    for (int j = 0; j < 4; ++j) o[j] = (v8f){};
    float mrow[8], lrow[8];
#pragma unroll
    for (int i = 0; i < 8; ++i) { mrow[i] = -__builtin_inff(); lrow[i] = 0.f; }

    for (int kb = 0; kb < SS; kb += 64) {
        // ---- scores S = (Q K^T) * 1/8 : 16 x 64, four 16x16 C-fragments ----
        v8f sf[4];
#pragma unroll
        for (int j = 0; j < 4; ++j) sf[j] = (v8f){};
#pragma unroll
        for (int kc = 0; kc < 2; ++kc)
#pragma unroll
            for (int j = 0; j < 4; ++j) {
                v16h kf = load_b_frag(Kb, DKK, kb + j * 16, kc * 32, lane);
                sf[j] = wmma16(qa[kc], kf, sf[j]);
            }
#pragma unroll
        for (int j = 0; j < 4; ++j)
#pragma unroll
            for (int i = 0; i < 8; ++i) sf[j][i] *= 0.125f;   // 1/sqrt(64)

        // ---- row max across 64 keys: elementwise then 16-lane xor-reduce ----
        float rmax[8];
#pragma unroll
        for (int i = 0; i < 8; ++i)
            rmax[i] = fmaxf(fmaxf(sf[0][i], sf[1][i]), fmaxf(sf[2][i], sf[3][i]));
#pragma unroll
        for (int i = 0; i < 8; ++i) {
            float v = rmax[i];
#pragma unroll
            for (int mask = 1; mask < 16; mask <<= 1)
                v = fmaxf(v, __shfl_xor(v, mask, 16));
            rmax[i] = v;
        }

        // ---- online softmax rescale ----
        float psum[8];
#pragma unroll
        for (int i = 0; i < 8; ++i) {
            const float mn    = fmaxf(mrow[i], rmax[i]);
            const float alpha = __expf(mrow[i] - mn);   // exp(-inf)=0 first iter
            mrow[i] = mn;
            lrow[i] *= alpha;
#pragma unroll
            for (int j = 0; j < 4; ++j) o[j][i] *= alpha;
            psum[i] = 0.f;
        }

        // ---- P = exp(S - m), stash to LDS in A-fragment-consumable layout ----
#pragma unroll
        for (int j = 0; j < 4; ++j)
#pragma unroll
            for (int i = 0; i < 8; ++i) {
                const float pp = __expf(sf[j][i] - mrow[i]);
                psum[i] += pp;
                pshare[w][hi * 8 + i][j * 16 + r] = (_Float16)pp;
            }
#pragma unroll
        for (int i = 0; i < 8; ++i) {
            float v = psum[i];
#pragma unroll
            for (int mask = 1; mask < 16; mask <<= 1)
                v += __shfl_xor(v, mask, 16);
            lrow[i] += v;
        }

        // ---- O += P * V  (per-wave LDS is in-order: no barrier needed) ----
        const _Float16* pb = &pshare[w][0][0];
#pragma unroll
        for (int kc = 0; kc < 2; ++kc) {
            v16h pa = load_a_frag(pb, 72, 0, kc * 32, lane);
#pragma unroll
            for (int j = 0; j < 4; ++j) {
                v16h vf = load_b_frag(Vb, SS, j * 16, kb + kc * 32, lane);
                o[j] = wmma16(pa, vf, o[j]);
            }
        }
    }

    // ---- finalize: O /= l, store merged-head f16 ctx [b][s][h*64+kd] ----
    const int bb = bh / HH, h = bh % HH;
#pragma unroll
    for (int j = 0; j < 4; ++j)
#pragma unroll
        for (int i = 0; i < 8; ++i) {
            const float val = o[j][i] / lrow[i];
            const int   s   = qbase + hi * 8 + i;
            ctx[(size_t)(bb * SS + s) * DMODEL + h * DKK + j * 16 + r] = (_Float16)val;
        }
}

// ---------------- host launcher ----------------
extern "C" void kernel_launch(void* const* d_in, const int* in_sizes, int n_in,
                              void* d_out, int out_size, void* d_ws, size_t ws_size,
                              hipStream_t stream) {
    (void)in_sizes; (void)n_in; (void)out_size; (void)ws_size;
    const float* q_in = (const float*)d_in[0];
    const float* k_in = (const float*)d_in[1];
    const float* v_in = (const float*)d_in[2];
    const float* Wq   = (const float*)d_in[3];
    const float* bq   = (const float*)d_in[4];
    const float* Wk   = (const float*)d_in[5];
    const float* bk   = (const float*)d_in[6];
    const float* Wv   = (const float*)d_in[7];
    const float* bv   = (const float*)d_in[8];
    const float* Wo   = (const float*)d_in[9];
    const float* bo   = (const float*)d_in[10];

    const int MSZ = MM * DMODEL;       // 3,145,728 elements
    const int WSZ = DMODEL * DMODEL;   //   589,824 elements

    char* p = (char*)d_ws;
    auto take = [&](size_t nbytes) -> char* {
        char* r = p;
        p += (nbytes + 255) & ~(size_t)255;
        return r;
    };
    _Float16* xq   = (_Float16*)take((size_t)MSZ * 2);
    _Float16* xk   = (_Float16*)take((size_t)MSZ * 2);
    _Float16* xv   = (_Float16*)take((size_t)MSZ * 2);
    _Float16* wq_h = (_Float16*)take((size_t)WSZ * 2);
    _Float16* wk_h = (_Float16*)take((size_t)WSZ * 2);
    _Float16* wv_h = (_Float16*)take((size_t)WSZ * 2);
    _Float16* wo_h = (_Float16*)take((size_t)WSZ * 2);
    _Float16* qh   = (_Float16*)take((size_t)MSZ * 2);  // [bh][s][64]
    _Float16* kh   = (_Float16*)take((size_t)MSZ * 2);  // [bh][s][64]
    _Float16* vT   = (_Float16*)take((size_t)MSZ * 2);  // [bh][64][s]
    _Float16* ctxh = (_Float16*)take((size_t)MSZ * 2);  // [b][s][768]

    auto cvt = [&](const float* s, _Float16* d, int n) {
        int blocks = (n + 1023) / 1024;
        if (blocks > 4096) blocks = 4096;
        mha_cvt_f32_f16<<<blocks, 256, 0, stream>>>(s, d, n);
    };
    cvt(q_in, xq, MSZ);
    cvt(k_in, xk, MSZ);
    cvt(v_in, xv, MSZ);
    cvt(Wq, wq_h, WSZ);
    cvt(Wk, wk_h, WSZ);
    cvt(Wv, wv_h, WSZ);
    cvt(Wo, wo_h, WSZ);

    dim3 gg(MM / 128, DMODEL / 128);   // 32 x 6
    mha_gemm_wmma<0><<<gg, 256, 0, stream>>>(xq, wq_h, bq, qh, MM, DMODEL, DMODEL);
    mha_gemm_wmma<0><<<gg, 256, 0, stream>>>(xk, wk_h, bk, kh, MM, DMODEL, DMODEL);
    mha_gemm_wmma<1><<<gg, 256, 0, stream>>>(xv, wv_h, bv, vT, MM, DMODEL, DMODEL);

    mha_flash_attn<<<dim3(SS / 128, BB * HH), 256, 0, stream>>>(qh, kh, vT, ctxh);

    mha_gemm_wmma<2><<<gg, 256, 0, stream>>>(ctxh, wo_h, bo, d_out, MM, DMODEL, DMODEL);
}
